// CorridorGenerationNetworkLSTM_755914244351
// MI455X (gfx1250) — compile-verified, matrix operand order checked by reading, not run on hardware
//
#include <hip/hip_runtime.h>

typedef __attribute__((ext_vector_type(16))) __bf16 v16bf;
typedef __attribute__((ext_vector_type(8)))  float  v8f;
typedef int v4i_gcc __attribute__((vector_size(16)));   // matches builtin's V4i param

#define T_STEPS 500
#define B_SZ    128
#define H_SZ    1024
#define DIN     576
#define NWG     32
#define NTHR    256

#if defined(__gfx1250__) && \
    __has_builtin(__builtin_amdgcn_global_load_async_to_lds_b128) && \
    __has_builtin(__builtin_amdgcn_s_wait_asynccnt)
#define HAVE_ASYNC_LDS 1
#else
#define HAVE_ASYNC_LDS 0
#endif

union FragBF { v16bf v; uint4 u[2]; };
union HU     { uint4 u; __bf16 h[8]; };

__device__ __forceinline__ float sigf(float x) { return 1.f / (1.f + __expf(-x)); }
__device__ __forceinline__ float tanhfast(float x) { return 2.f / (1.f + __expf(-2.f * x)) - 1.f; }

// ---------------- prep kernels ----------------

// x = concat(pcl_features, state @ W_state + b_state)   [128 x 576]
__global__ void build_x_kern(const float* __restrict__ pcl, const float* __restrict__ st,
                             const float* __restrict__ Ws, const float* __restrict__ bs,
                             float* __restrict__ x) {
  int idx = blockIdx.x * NTHR + threadIdx.x;
  if (idx >= B_SZ * DIN) return;
  int b = idx / DIN, j = idx % DIN;
  float v;
  if (j < 512) {
    v = pcl[b * 512 + j];
  } else {
    int s = j - 512;
    v = bs[s];
    #pragma unroll
    for (int k = 0; k < 9; ++k) v += st[b * 9 + k] * Ws[k * 64 + s];
  }
  x[idx] = v;
}

// x_proj = x @ W_ih^T + b_ih + b_hh, written directly in WMMA C-fragment order:
// [wg][mt][nt][lane][r]  (r = accumulator VGPR index 0..7)
__global__ void xproj_swz_kern(const float* __restrict__ x, const float* __restrict__ Wih,
                               const float* __restrict__ bih, const float* __restrict__ bhh,
                               float* __restrict__ xpswz) {
  __shared__ float xs[DIN];
  int b = blockIdx.x >> 4;
  int r = (blockIdx.x & 15) * NTHR + threadIdx.x;   // gate row 0..4095
  for (int i = threadIdx.x; i < DIN; i += NTHR) xs[i] = x[b * DIN + i];
  __syncthreads();
  float v = bih[r] + bhh[r];
  const float* wr = Wih + (size_t)r * DIN;
  for (int k = 0; k < DIN; ++k) v += xs[k] * wr[k];
  // map (m=b, col=r) -> swizzled C-fragment position
  int g = r >> 10, rem = r & 1023, wg = rem >> 5, lcl = rem & 31;
  int pair = lcl >> 4, n = lcl & 15;
  int nt = g * 2 + pair;
  int mt = b >> 4, mr = b & 15, rr = mr & 7, hm = mr >> 3;
  int ln = hm * 16 + n;
  xpswz[(((((size_t)wg * 8 + mt) * 8 + nt) * 32 + ln) * 8 + rr)] = v;
}

// W_hh (f32, [4096x1024]) -> bf16 B-fragments: [wg][nt][kt][lane][16 bf16]
__global__ void whh_swz_kern(const float* __restrict__ Whh, __bf16* __restrict__ wswz) {
  int o = blockIdx.x * NTHR + threadIdx.x;          // 0 .. 4194303
  int e = o & 15, ln = (o >> 4) & 31, kt = (o >> 9) & 31, nt = (o >> 14) & 7, wg = o >> 17;
  int n = ln & 15, hf = ln >> 4;
  int kl = (e < 8 ? e : e + 8) + hf * 8;            // ISA 16-bit operand K layout
  int k = kt * 32 + kl;
  int row = (nt >> 1) * 1024 + wg * 32 + (nt & 1) * 16 + n;  // gate-row owned by wg
  wswz[o] = (__bf16)Whh[(size_t)row * H_SZ + k];
}

__global__ void init_kern(__bf16* __restrict__ h0, unsigned* __restrict__ ctr) {
  int i = blockIdx.x * NTHR + threadIdx.x;
  if (i < B_SZ * H_SZ) h0[i] = (__bf16)0.f;
  if (i == 0) *ctr = 0u;
}

// ---------------- persistent LSTM kernel ----------------
__global__ void __launch_bounds__(NTHR, 1)
lstm_main(const __bf16* __restrict__ wswz, const float* __restrict__ xpswz,
          __bf16* __restrict__ hb0, __bf16* __restrict__ hb1,
          unsigned* __restrict__ ctr,
          const float* __restrict__ Wel, const float* __restrict__ bel,
          const float* __restrict__ Wee, const float* __restrict__ bee,
          const float* __restrict__ Wpe, const float* __restrict__ bpe,
          float* __restrict__ out) {
  extern __shared__ uint4 smem4[];                  // 256 KB: resident W_hh slice (bf16)
  char* smem = (char*)smem4;
  const int wg = blockIdx.x;
  const int tid = threadIdx.x;
  const int wv = tid >> 5;
  const int lane = tid & 31;

  // one-time fill of LDS with this WGP's pre-swizzled weight slice
  {
    const uint4* src = (const uint4*)(wswz + (size_t)wg * (8 * 32 * 32 * 16));
#if HAVE_ASYNC_LDS
    for (int i = tid; i < 16384; i += NTHR) {
      __builtin_amdgcn_global_load_async_to_lds_b128(
          (__attribute__((address_space(1))) v4i_gcc*)(src + i),
          (__attribute__((address_space(3))) v4i_gcc*)(smem4 + i), 0, 0);
    }
    __builtin_amdgcn_s_wait_asynccnt(0);
#else
    for (int i = tid; i < 16384; i += NTHR) smem4[i] = src[i];
#endif
  }
  __syncthreads();

  const int mt = wv;                                // wave's batch tile (16 rows)
  const int mrow = mt * 16 + (lane & 15);
  const int halfk = lane >> 4;
  const float* xp = xpswz + (size_t)((wg * 8 + mt) * 8) * (32 * 8);

  float cst[16];
  #pragma unroll
  for (int i = 0; i < 16; ++i) cst[i] = 0.f;

  for (int t = 0; t < T_STEPS; ++t) {
    const __bf16* hin  = (t & 1) ? hb1 : hb0;
    __bf16*       hout = (t & 1) ? hb0 : hb1;

    // init accumulators with x_proj (+biases), pre-swizzled to C layout
    v8f acc[8];
    #pragma unroll
    for (int nt = 0; nt < 8; ++nt)
      acc[nt] = *(const v8f*)(xp + (nt * 32 + lane) * 8);

    // gates[16 x 128] += h[16 x 1024] @ Whh_slice^T  via bf16 WMMA
    const __bf16* arow = hin + (size_t)mrow * H_SZ + halfk * 8;
    for (int kt = 0; kt < 32; ++kt) {
      FragBF a;
      a.u[0] = *(const uint4*)(arow + kt * 32);
      a.u[1] = *(const uint4*)(arow + kt * 32 + 16);
      // preload ALL 8 B fragments first: one ds clause + single wait,
      // then 8 back-to-back WMMAs (independent acc chains, distinct B regs)
      FragBF b[8];
      #pragma unroll
      for (int nt = 0; nt < 8; ++nt) {
        const uint4* bp = (const uint4*)(smem + (size_t)((nt * 32 + kt) * 32 + lane) * 32);
        b[nt].u[0] = bp[0];
        b[nt].u[1] = bp[1];
      }
      #pragma unroll
      for (int nt = 0; nt < 8; ++nt)
        acc[nt] = __builtin_amdgcn_wmma_f32_16x16x32_bf16(
            false, a.v, false, b[nt].v, (short)0, acc[nt], false, false);
    }

    // LSTM cell update: i,f,g,o live in acc[pair], acc[2+pair], acc[4+pair], acc[6+pair]
    #pragma unroll
    for (int pair = 0; pair < 2; ++pair) {
      const int ucol = wg * 32 + pair * 16 + (lane & 15);
      #pragma unroll
      for (int r = 0; r < 8; ++r) {
        float ig = sigf(acc[0 + pair][r]);
        float fg = sigf(acc[2 + pair][r]);
        float gg = tanhfast(acc[4 + pair][r]);
        float og = sigf(acc[6 + pair][r]);
        float c  = fg * cst[pair * 8 + r] + ig * gg;
        cst[pair * 8 + r] = c;
        float h  = og * tanhfast(c);
        int m = mt * 16 + r + 8 * halfk;
        hout[(size_t)m * H_SZ + ucol] = (__bf16)h;
      }
    }

    // grid-wide step barrier (monotonic counter, double-buffered h)
    __threadfence();
    __syncthreads();
    if (tid == 0) {
      __hip_atomic_fetch_add(ctr, 1u, __ATOMIC_RELEASE, __HIP_MEMORY_SCOPE_AGENT);
      const unsigned tgt = (unsigned)(t + 1) * NWG;
      while (__hip_atomic_load(ctr, __ATOMIC_ACQUIRE, __HIP_MEMORY_SCOPE_AGENT) < tgt)
        __builtin_amdgcn_s_sleep(1);
    }
    __syncthreads();

    // output heads: this WG handles batches wg*4..wg*4+3 (24 dot products / 8 waves)
    #pragma unroll
    for (int q = 0; q < 3; ++q) {
      int task = wv * 3 + q;
      int bl = task / 6, oi = task % 6;
      int m = wg * 4 + bl;
      const __bf16* hr = hout + (size_t)m * H_SZ;
      const int kbase = lane * 32;                  // blocked 32 elements per lane
      float s = 0.f;
      #pragma unroll
      for (int j = 0; j < 4; ++j) {
        HU hu;
        hu.u = *(const uint4*)(hr + kbase + j * 8);
        #pragma unroll
        for (int e = 0; e < 8; ++e) {
          int k = kbase + j * 8 + e;
          float wt = (oi < 4) ? Wel[k * 4 + oi] : (oi == 4 ? Wee[k] : Wpe[k]);
          s += (float)hu.h[e] * wt;
        }
      }
      #pragma unroll
      for (int off = 16; off > 0; off >>= 1) s += __shfl_xor(s, off, 32);
      if (lane == 0) {
        if (oi < 4)
          out[((size_t)m * T_STEPS + t) * 4 + oi] = s + bel[oi];
        else if (oi == 4)
          out[256000 + (size_t)m * T_STEPS + t] = sigf(s + bee[0]);
        else
          out[320000 + (size_t)m * T_STEPS + t] = sigf(s + bpe[0]);
      }
    }
  }
}

// ---------------- launcher ----------------
extern "C" void kernel_launch(void* const* d_in, const int* in_sizes, int n_in,
                              void* d_out, int out_size, void* d_ws, size_t ws_size,
                              hipStream_t stream) {
  (void)in_sizes; (void)n_in; (void)out_size; (void)ws_size;
  const float* pcl = (const float*)d_in[0];
  const float* st  = (const float*)d_in[1];
  const float* Ws  = (const float*)d_in[2];
  const float* bs  = (const float*)d_in[3];
  const float* Wih = (const float*)d_in[4];
  const float* bih = (const float*)d_in[5];
  const float* Whh = (const float*)d_in[6];
  const float* bhh = (const float*)d_in[7];
  const float* Wel = (const float*)d_in[8];
  const float* bel = (const float*)d_in[9];
  const float* Wee = (const float*)d_in[10];
  const float* bee = (const float*)d_in[11];
  const float* Wpe = (const float*)d_in[12];
  const float* bpe = (const float*)d_in[13];
  float* out = (float*)d_out;

  char* ws = (char*)d_ws;
  unsigned* ctr  = (unsigned*)ws;                         // 256 B slot
  float*  x      = (float*)(ws + 256);                    // 294912 B
  float*  xpswz  = (float*)(ws + 295168);                 // 2 MB
  __bf16* wswz   = (__bf16*)(ws + 2392320);               // 8 MB
  __bf16* hb0    = (__bf16*)(ws + 10780928);              // 256 KB
  __bf16* hb1    = (__bf16*)(ws + 11043072);              // 256 KB  (total ~11.3 MB)

  hipLaunchKernelGGL(init_kern,      dim3(512),   dim3(NTHR), 0, stream, hb0, ctr);
  hipLaunchKernelGGL(build_x_kern,   dim3(288),   dim3(NTHR), 0, stream, pcl, st, Ws, bs, x);
  hipLaunchKernelGGL(xproj_swz_kern, dim3(2048),  dim3(NTHR), 0, stream, x, Wih, bih, bhh, xpswz);
  hipLaunchKernelGGL(whh_swz_kern,   dim3(16384), dim3(NTHR), 0, stream, Whh, wswz);
  hipLaunchKernelGGL(lstm_main, dim3(NWG), dim3(NTHR), 262144, stream,
                     wswz, xpswz, hb0, hb1, ctr, Wel, bel, Wee, bee, Wpe, bpe, out);
}
